// AttnProcessor_81905026335371
// MI455X (gfx1250) — matrix operand order repaired and checked by reading, not compiled, not run on hardware
//
#include <hip/hip_runtime.h>
#include <hip/hip_bf16.h>

// ---------------------------------------------------------------------------
// Attention (B=2, S=4096, D=512, H=8, HD=64) for gfx1250 (MI455X).
// bf16 WMMA (v_wmma_f32_16x16x32_bf16) with fp32 accumulate, flash-style
// fused softmax, CDNA5 async global->LDS staging (double-buffered,
// global_load_async_to_lds_b128 + s_wait_asynccnt) and global_prefetch.
// All transposes hoisted out of inner loops so every LDS fragment load is a
// contiguous dword path (compiler merges to ds_load_b128).
// ---------------------------------------------------------------------------

#define Bsz 2
#define Ssz 4096
#define Dsz 512
#define Hsz 8
#define HDsz 64
#define Msz (Bsz * Ssz)   // 8192

typedef __attribute__((ext_vector_type(16))) __bf16         v16bf;
typedef __attribute__((ext_vector_type(8)))  float          v8f;
typedef __attribute__((ext_vector_type(8)))  unsigned int   v8u32;

// ---- helpers ---------------------------------------------------------------

__device__ __forceinline__ unsigned short f2bf(float f) {
  unsigned int u = __builtin_bit_cast(unsigned int, f);
  return (unsigned short)((u + 0x7FFFu + ((u >> 16) & 1u)) >> 16);  // RNE
}

__device__ __forceinline__ v8f wmma_bf16(v16bf a, v16bf b, v8f c) {
  // D = A(16x32 bf16) * B(32x16 bf16) + C(16x16 f32)
  return __builtin_amdgcn_wmma_f32_16x16x32_bf16(
      false, a, false, b, (short)0, c, false, false);
}

// CDNA5 async copy: 16B global -> LDS per lane, tracked by ASYNCcnt.
__device__ __forceinline__ void async_load_b128(unsigned lds_off,
                                                const void* gaddr) {
  asm volatile("global_load_async_to_lds_b128 %0, %1, off"
               :: "v"(lds_off), "v"(gaddr) : "memory");
}
__device__ __forceinline__ void wait_async0() {
  asm volatile("s_wait_asynccnt 0x0" ::: "memory");
}
__device__ __forceinline__ unsigned lds_off32(const void* p) {
  return (unsigned)(uintptr_t)p;  // low 32 bits of generic ptr = LDS offset
}

// A-fragment (16x32 bf16). lanes 0-15: M=lane; VGPR v: K={2v,2v+1}(+8 if v>=4)(+8*half).
// lds row-major [row][rowStride], K contiguous -> dword loads (merge to b128).
__device__ __forceinline__ v16bf load_a_frag(const unsigned short* lds, int row0,
                                             int rowStride, int kOff, int lane) {
  const int half = (lane >> 4) & 1;
  const int m = lane & 15;
  const unsigned short* base = lds + (row0 + m) * rowStride + kOff;
  v8u32 f;
#pragma unroll
  for (int v = 0; v < 8; ++v) {
    int kb = (v < 4 ? 2 * v : 16 + 2 * (v - 4)) + half * 8;
    f[v] = *reinterpret_cast<const unsigned int*>(base + kb);
  }
  return __builtin_bit_cast(v16bf, f);
}

// B-fragment (32x16 bf16): lane n = column; VGPR v: K={2v,2v+1}(+8 if v>=4)(+8*half), same as A.
// lds laid out [col][k] (reduction dim contiguous) -> dword loads.
__device__ __forceinline__ v16bf load_b_frag(const unsigned short* lds, int col0,
                                             int rowStride, int kOff, int lane) {
  const int half = (lane >> 4) & 1;
  const int n = lane & 15;
  const unsigned short* base = lds + (col0 + n) * rowStride + kOff;
  v8u32 f;
#pragma unroll
  for (int v = 0; v < 8; ++v) {
    int kb = (v < 4 ? 2 * v : 16 + 2 * (v - 4)) + half * 8;
    f[v] = *reinterpret_cast<const unsigned int*>(base + kb);
  }
  return __builtin_bit_cast(v16bf, f);
}

// ---- fp32 -> bf16 conversion (plain and transposing) -----------------------

__global__ __launch_bounds__(256) void cvt_f32_bf16_kernel(
    const float* __restrict__ in, unsigned short* __restrict__ out, int n) {
  int i = blockIdx.x * blockDim.x + threadIdx.x;
  if (i < n) out[i] = f2bf(in[i]);
}

// out[n][k] = bf16(in[k][n]) for 512x512 weights: B fragments become
// contiguous in LDS for every GEMM tile.
__global__ __launch_bounds__(256) void cvt_f32_bf16_T_kernel(
    const float* __restrict__ in, unsigned short* __restrict__ out) {
  int i = blockIdx.x * blockDim.x + threadIdx.x;  // i = n*512 + k
  int n = i >> 9, k = i & 511;
  out[i] = f2bf(in[k * Dsz + n]);
}

// ---- bf16 GEMM: out[M,512] = A[M,512] @ W[512,512], epilogue (+bias)*scale -
// Wt is the pre-transposed weight [n][k]. 128 threads (4 waves), block tile
// 128x64; wave tile 32x64 (2 A-frags x 4 B-frags = 8 WMMA / 32-deep K step).
// Double-buffered async LDS pipeline: tile k+1 DMA overlaps tile k WMMAs.
// writeMode: 0 = bf16 [M][512]; 1 = f32 [M][512]; 2 = bf16 [b][h][d][s] (V).

__global__ __launch_bounds__(128) void gemm_bf16_kernel(
    const unsigned short* __restrict__ A, const unsigned short* __restrict__ Wt,
    const float* __restrict__ bias, unsigned short* __restrict__ outB,
    float* __restrict__ outF, int writeMode, float scale) {
  __shared__ unsigned short lA[2][128 * 32];  // A tile row-major [128][32]
  __shared__ unsigned short lW[2][64 * 32];   // Wt tile row-major [n][32]

  const int tid = threadIdx.x;
  const int lane = tid & 31;
  const int wave = tid >> 5;
  const int n0 = blockIdx.x * 64;
  const int m0 = blockIdx.y * 128;

  v8f acc[2][4];
#pragma unroll
  for (int mi = 0; mi < 2; ++mi)
#pragma unroll
    for (int t = 0; t < 4; ++t)
#pragma unroll
      for (int r = 0; r < 8; ++r) acc[mi][t][r] = 0.0f;

  // 6 async b128 per thread per tile (A: 4, W: 2).
  auto issue_tile = [&](int kk, int buf) {
    const int k0 = kk * 32;
#pragma unroll
    for (int i = tid; i < 512; i += 128) {
      int r = i >> 2, cv = i & 3;
      async_load_b128(lds_off32(&lA[buf][r * 32 + cv * 8]),
                      A + (size_t)(m0 + r) * Dsz + k0 + cv * 8);
    }
#pragma unroll
    for (int i = tid; i < 256; i += 128) {
      int n = i >> 1, cv = i & 1;
      async_load_b128(lds_off32(&lW[buf][n * 32 + cv * 16]),
                      Wt + (size_t)(n0 + n) * Dsz + k0 + cv * 16);
    }
  };

  issue_tile(0, 0);
  wait_async0();
  __syncthreads();

  for (int kk = 0; kk < 16; ++kk) {
    const int buf = kk & 1;
    if (kk + 1 < 16) issue_tile(kk + 1, buf ^ 1);  // overlap DMA with WMMA

    v16bf a0 = load_a_frag(lA[buf], wave * 32, 32, 0, lane);
    v16bf a1 = load_a_frag(lA[buf], wave * 32 + 16, 32, 0, lane);
#pragma unroll
    for (int t = 0; t < 4; ++t) {
      v16bf b = load_b_frag(lW[buf], t * 16, 32, 0, lane);
      acc[0][t] = wmma_bf16(a0, b, acc[0][t]);
      acc[1][t] = wmma_bf16(a1, b, acc[1][t]);
    }

    wait_async0();      // next tile landed
    __syncthreads();    // and everyone is done reading this tile
  }

  // Epilogue. C layout: VGPR r -> row r + 8*(lane>=16), col = lane&15 (+16t).
  const int half = lane >> 4;
  const int nn = lane & 15;
#pragma unroll
  for (int mi = 0; mi < 2; ++mi) {
#pragma unroll
    for (int t = 0; t < 4; ++t) {
#pragma unroll
      for (int r = 0; r < 8; ++r) {
        int row = m0 + wave * 32 + mi * 16 + r + half * 8;
        int col = n0 + t * 16 + nn;
        float v = (acc[mi][t][r] + bias[col]) * scale;
        if (writeMode == 1) {
          outF[(size_t)row * Dsz + col] = v;
        } else if (writeMode == 2) {
          // V: store head-transposed [b][h][d][s] so attention PV fragments
          // are contiguous in LDS.
          size_t idx = (((size_t)(row >> 12) * Hsz + (col >> 6)) * HDsz +
                        (col & 63)) * Ssz + (row & 4095);
          outB[idx] = f2bf(v);
        } else {
          outB[(size_t)row * Dsz + col] = f2bf(v);
        }
      }
    }
  }
}

// ---- fused flash attention -------------------------------------------------
// Grid: (S/64, B*H). 128 threads (4 waves); wave owns 16 query rows x HD=64.
// Q pre-scaled by 1/sqrt(HD). K tiles [key][d]; V tiles [d][key] (from the
// head-transposed V buffer) -> all fragments contiguous. Double-buffered
// async K/V staging.

__global__ __launch_bounds__(128) void flash_attn_kernel(
    const unsigned short* __restrict__ Qb, const unsigned short* __restrict__ Kb,
    const unsigned short* __restrict__ VbT, unsigned short* __restrict__ Ob) {
  __shared__ unsigned short lQ[64 * 64];       // [q][d]
  __shared__ unsigned short lK[2][64 * 64];    // [key][d] == B^T for scores
  __shared__ unsigned short lV[2][64 * 64];    // [d][key] (pre-transposed)
  __shared__ unsigned short lP[64 * 64];       // probs staging (C->A relayout)

  const int tid = threadIdx.x;
  const int lane = tid & 31;
  const int wave = tid >> 5;
  const int half = lane >> 4;
  const int nn = lane & 15;

  const int q0 = blockIdx.x * 64;
  const int bb = blockIdx.y >> 3;
  const int h = blockIdx.y & 7;
  const size_t rowBase = (size_t)bb * Ssz + q0;            // into [M]
  const int hcol = h * HDsz;
  const unsigned short* Vhead =
      VbT + (((size_t)bb * Hsz + h) * HDsz) * Ssz;         // [d][s]

  // 8 async b128 per thread per KV tile (K: 4, V: 4).
  auto issue_tile = [&](int kt, int buf) {
    const size_t kbase = (size_t)bb * Ssz + kt * 64;
#pragma unroll
    for (int i = tid; i < 512; i += 128) {
      int r = i >> 3, cv = i & 7;
      async_load_b128(lds_off32(&lK[buf][r * 64 + cv * 8]),
                      Kb + (kbase + r) * Dsz + hcol + cv * 8);
      async_load_b128(lds_off32(&lV[buf][r * 64 + cv * 8]),
                      Vhead + (size_t)r * Ssz + kt * 64 + cv * 8);
    }
  };

  // Stage Q once + first KV tile.
#pragma unroll
  for (int i = tid; i < 512; i += 128) {
    int r = i >> 3, cv = i & 7;
    async_load_b128(lds_off32(&lQ[r * 64 + cv * 8]),
                    Qb + (rowBase + r) * Dsz + hcol + cv * 8);
  }
  issue_tile(0, 0);
  wait_async0();
  __syncthreads();

  v16bf qa0 = load_a_frag(lQ, wave * 16, 64, 0, lane);
  v16bf qa1 = load_a_frag(lQ, wave * 16, 64, 32, lane);

  v8f oacc[4];
#pragma unroll
  for (int t = 0; t < 4; ++t)
#pragma unroll
    for (int r = 0; r < 8; ++r) oacc[t][r] = 0.0f;
  float mrow[8], lrow[8];
#pragma unroll
  for (int r = 0; r < 8; ++r) { mrow[r] = -1e30f; lrow[r] = 0.0f; }

  for (int kt = 0; kt < Ssz / 64; ++kt) {
    const int buf = kt & 1;
    if (kt + 1 < Ssz / 64) issue_tile(kt + 1, buf ^ 1);  // overlap DMA
    if (kt + 2 < Ssz / 64) {                             // L2 prefetch kt+2
      const size_t nb = (size_t)bb * Ssz + (kt + 2) * 64 + (tid >> 1);
      __builtin_prefetch(Kb + nb * Dsz + hcol + (tid & 1) * 32, 0, 1);
    }

    // Scores: S(16x64) = Q @ K^T, two 32-deep chunks over HD.
    v8f sacc[4];
#pragma unroll
    for (int t = 0; t < 4; ++t)
#pragma unroll
      for (int r = 0; r < 8; ++r) sacc[t][r] = 0.0f;
#pragma unroll
    for (int t = 0; t < 4; ++t)
      sacc[t] = wmma_bf16(qa0, load_b_frag(lK[buf], t * 16, 64, 0, lane),
                          sacc[t]);
#pragma unroll
    for (int t = 0; t < 4; ++t)
      sacc[t] = wmma_bf16(qa1, load_b_frag(lK[buf], t * 16, 64, 32, lane),
                          sacc[t]);

    // Online softmax (fp32). C row spans 16 lanes -> 16-wide xor reductions.
#pragma unroll
    for (int r = 0; r < 8; ++r) {
      float mx = fmaxf(fmaxf(sacc[0][r], sacc[1][r]),
                       fmaxf(sacc[2][r], sacc[3][r]));
#pragma unroll
      for (int off = 8; off >= 1; off >>= 1)
        mx = fmaxf(mx, __shfl_xor(mx, off, 32));
      float mnew = fmaxf(mrow[r], mx);
      float corr = __expf(mrow[r] - mnew);
      float rs = 0.0f;
#pragma unroll
      for (int t = 0; t < 4; ++t) {
        float p = __expf(sacc[t][r] - mnew);
        sacc[t][r] = p;
        rs += p;
      }
#pragma unroll
      for (int off = 8; off >= 1; off >>= 1) rs += __shfl_xor(rs, off, 32);
      lrow[r] = lrow[r] * corr + rs;
      mrow[r] = mnew;
#pragma unroll
      for (int t = 0; t < 4; ++t) oacc[t][r] *= corr;
    }

    // Relayout P (C-layout) -> A-layout through LDS as bf16.
#pragma unroll
    for (int t = 0; t < 4; ++t)
#pragma unroll
      for (int r = 0; r < 8; ++r)
        lP[(wave * 16 + r + half * 8) * 64 + t * 16 + nn] = f2bf(sacc[t][r]);
    __syncthreads();

    v16bf pa0 = load_a_frag(lP, wave * 16, 64, 0, lane);
    v16bf pa1 = load_a_frag(lP, wave * 16, 64, 32, lane);
#pragma unroll
    for (int t = 0; t < 4; ++t)
      oacc[t] = wmma_bf16(pa0, load_b_frag(lV[buf], t * 16, 64, 0, lane),
                          oacc[t]);
#pragma unroll
    for (int t = 0; t < 4; ++t)
      oacc[t] = wmma_bf16(pa1, load_b_frag(lV[buf], t * 16, 64, 32, lane),
                          oacc[t]);

    wait_async0();      // tile kt+1 landed
    __syncthreads();    // all waves done with tile kt (and lP)
  }

  // 1/l normalization; write back into [B,S,D] (head-interleaved) layout.
#pragma unroll
  for (int t = 0; t < 4; ++t) {
#pragma unroll
    for (int r = 0; r < 8; ++r) {
      int row = wave * 16 + r + half * 8;
      float v = oacc[t][r] / lrow[r];
      Ob[(rowBase + row) * Dsz + hcol + t * 16 + nn] = f2bf(v);
    }
  }
}

// ---- launcher --------------------------------------------------------------

extern "C" void kernel_launch(void* const* d_in, const int* in_sizes, int n_in,
                              void* d_out, int out_size, void* d_ws, size_t ws_size,
                              hipStream_t stream) {
  (void)in_sizes; (void)n_in; (void)out_size; (void)ws_size;
  const float* x  = (const float*)d_in[0];
  const float* Wq = (const float*)d_in[1];
  const float* bq = (const float*)d_in[2];
  const float* Wk = (const float*)d_in[3];
  const float* bk = (const float*)d_in[4];
  const float* Wv = (const float*)d_in[5];
  const float* bv = (const float*)d_in[6];
  const float* Wo = (const float*)d_in[7];
  const float* bo = (const float*)d_in[8];
  float* out = (float*)d_out;

  const size_t MD = (size_t)Msz * Dsz;   // 4,194,304 elems
  const size_t DD = (size_t)Dsz * Dsz;   // 262,144 elems
  unsigned short* ws  = (unsigned short*)d_ws;
  unsigned short* xb  = ws;              // bf16 x
  unsigned short* wqb = xb + MD;         // transposed bf16 weights [n][k]
  unsigned short* wkb = wqb + DD;
  unsigned short* wvb = wkb + DD;
  unsigned short* wob = wvb + DD;
  unsigned short* qb  = wob + DD;        // Q (pre-scaled by 1/sqrt(HD))
  unsigned short* kb  = qb + MD;
  unsigned short* vb  = kb + MD;         // V head-transposed [b][h][d][s]
  unsigned short* ob  = vb + MD;         // attention output (pre-projection)

  cvt_f32_bf16_kernel<<<(MD + 255) / 256, 256, 0, stream>>>(x, xb, (int)MD);
  cvt_f32_bf16_T_kernel<<<(DD + 255) / 256, 256, 0, stream>>>(Wq, wqb);
  cvt_f32_bf16_T_kernel<<<(DD + 255) / 256, 256, 0, stream>>>(Wk, wkb);
  cvt_f32_bf16_T_kernel<<<(DD + 255) / 256, 256, 0, stream>>>(Wv, wvb);
  cvt_f32_bf16_T_kernel<<<(DD + 255) / 256, 256, 0, stream>>>(Wo, wob);

  dim3 gg(Dsz / 64, Msz / 128);  // (8, 64)
  gemm_bf16_kernel<<<gg, 128, 0, stream>>>(xb, wqb, bq, qb, nullptr, 0, 0.125f);
  gemm_bf16_kernel<<<gg, 128, 0, stream>>>(xb, wkb, bk, kb, nullptr, 0, 1.0f);
  gemm_bf16_kernel<<<gg, 128, 0, stream>>>(xb, wvb, bv, vb, nullptr, 2, 1.0f);

  flash_attn_kernel<<<dim3(Ssz / 64, Bsz * Hsz), 128, 0, stream>>>(qb, kb, vb, ob);

  gemm_bf16_kernel<<<gg, 128, 0, stream>>>(ob, wob, bo, nullptr, out, 1, 1.0f);
}
